// TDrumorGCN_9354438770817
// MI455X (gfx1250) — matrix operand order repaired and checked by reading, not compiled, not run on hardware
//
#include <hip/hip_runtime.h>
#include <hip/hip_bf16.h>

typedef __attribute__((ext_vector_type(16))) _Float16 v16h;
typedef __attribute__((ext_vector_type(8)))  _Float16 v8h;
typedef __attribute__((ext_vector_type(4)))  _Float16 v4h;
typedef __attribute__((ext_vector_type(8)))  float    v8f;

#define HID 64
#define IN1 128
#define INF_ 256   // IN = 256
#define LDSTR 40   // LDS row stride in halves (multiple of 8 -> aligned 16B chunks)

// ---------------- utility ----------------
__global__ void fillf_kernel(float* __restrict__ p, float v, int n) {
    int i = blockIdx.x * blockDim.x + threadIdx.x;
    if (i < n) p[i] = v;
}

__global__ void deg_kernel(const int* __restrict__ ei, float* __restrict__ deg, int E) {
    int e = blockIdx.x * blockDim.x + threadIdx.x;
    if (e < E) atomicAdd(&deg[ei[E + e]], 1.0f);
}

__global__ void dinv_kernel(const float* __restrict__ deg, float* __restrict__ dinv, int N) {
    int i = blockIdx.x * blockDim.x + threadIdx.x;
    if (i < N) dinv[i] = rsqrtf(deg[i]);
}

// ---- pack a [K x 64] f32 weight slab into f16 WMMA B-fragment order ----
__global__ void packW_kernel(const float* __restrict__ W, _Float16* __restrict__ outp,
                             int ktiles, int rowoff) {
    int t = blockIdx.x * blockDim.x + threadIdx.x;
    int total = ktiles * 4 * 512;
    if (t >= total) return;
    int e    = t & 15;
    int lane = (t >> 4) & 31;
    int frag = t >> 9;
    int nt = frag & 3, ktile = frag >> 2;
    int half = lane >> 4, n = lane & 15;
    int k = ktile * 32 + 8 * half + e + (e >= 8 ? 8 : 0);
    outp[t] = (_Float16)W[(size_t)(rowoff + k) * HID + nt * 16 + n];
}

__device__ inline v16h join16(v8h a0, v8h a1) {
    v16h a;
    #pragma unroll
    for (int e = 0; e < 8; ++e) { a[e] = a0[e]; a[e + 8] = a1[e]; }
    return a;
}

__device__ inline v4h cvt4(float4 f) {
    v4h h;
    h[0] = (_Float16)f.x; h[1] = (_Float16)f.y;
    h[2] = (_Float16)f.z; h[3] = (_Float16)f.w;
    return h;
}

// ---------------- GEMM1: h1 = [x|x_da] @ W1  (K=256, LDS-staged A) ----------------
__global__ void gemm1_wmma(const float* __restrict__ x, const float* __restrict__ xda,
                           const _Float16* __restrict__ Wp, float* __restrict__ h1, int N) {
    __shared__ _Float16 As[128 * LDSTR];
    int tid = threadIdx.x;
    int wave = tid >> 5, lane = tid & 31;
    int half = lane >> 4, mr = lane & 15, ncol = lane & 15;
    int blockRow0 = blockIdx.x * 128;

    v8f acc[4] = {v8f{}, v8f{}, v8f{}, v8f{}};
    for (int kt = 0; kt < INF_; kt += 32) {
        const float* base = (kt < IN1) ? (x + kt) : (xda + (kt - IN1));
        __syncthreads();
        // stage 128 rows x 32 cols: 1024 float4 groups, 4 per thread
        float4 f[4]; int r_[4], c4_[4];
        #pragma unroll
        for (int i = 0; i < 4; ++i) {
            int g = tid + i * 256;
            r_[i] = g >> 3; c4_[i] = (g & 7) * 4;
            int rg = blockRow0 + r_[i]; if (rg >= N) rg = N - 1;
            f[i] = *(const float4*)(&base[(size_t)rg * IN1 + c4_[i]]);
        }
        #pragma unroll
        for (int i = 0; i < 4; ++i)
            *(v4h*)(&As[r_[i] * LDSTR + c4_[i]]) = cvt4(f[i]);
        __syncthreads();
        const _Float16* ap = &As[(wave * 16 + mr) * LDSTR + 8 * half];
        v16h a = join16(*(const v8h*)ap, *(const v8h*)(ap + 16));
        int fb = (kt >> 5) * 4;
        #pragma unroll
        for (int nt = 0; nt < 4; ++nt) {
            v16h b = *(const v16h*)(Wp + (size_t)(fb + nt) * 512 + lane * 16);
            acc[nt] = __builtin_amdgcn_wmma_f32_16x16x32_f16(
                false, a, false, b, (short)0, acc[nt], false, false);
        }
    }
    int m0 = blockRow0 + wave * 16;
    #pragma unroll
    for (int v = 0; v < 8; ++v) {
        int r = m0 + v + 8 * half;
        if (r < N) {
            float* o = h1 + (size_t)r * HID + ncol;
            o[0]  = acc[0][v];
            o[16] = acc[1][v];
            o[32] = acc[2][v];
            o[48] = acc[3][v];
        }
    }
}

// ------- root GEMM: g2[b] = relu(x0[rootindex[b]]) @ W2[64:320]  (M=128, K=256) -------
__global__ void rootgemm_wmma(const float* __restrict__ x, const float* __restrict__ xda,
                              const _Float16* __restrict__ WpB, const int* __restrict__ rootindex,
                              float* __restrict__ g2, int B) {
    int gw = blockIdx.x * 8 + (threadIdx.x >> 5);   // 0..31
    int lane = threadIdx.x & 31;
    int mt = gw >> 2, nt = gw & 3;
    int m0 = mt * 16;
    if (m0 >= B) return;
    int half = lane >> 4;
    int rr = m0 + (lane & 15); if (rr >= B) rr = B - 1;
    int node = rootindex[rr];
    const float* xr  = x   + (size_t)node * IN1;
    const float* xdr = xda + (size_t)node * IN1;
    int ncol = lane & 15;

    v8f acc = {};
    for (int kt = 0; kt < INF_; kt += 32) {
        // fragment k-pattern = two contiguous 8-float runs; kt%32==0 so no x/xda split
        const float* src = (kt < IN1) ? (xr + kt) : (xdr + (kt - IN1));
        float4 f0 = *(const float4*)(src + 8 * half);
        float4 f1 = *(const float4*)(src + 8 * half + 4);
        float4 f2 = *(const float4*)(src + 16 + 8 * half);
        float4 f3 = *(const float4*)(src + 16 + 8 * half + 4);
        v16h a;
        a[0]  = (_Float16)fmaxf(f0.x, 0.f); a[1]  = (_Float16)fmaxf(f0.y, 0.f);
        a[2]  = (_Float16)fmaxf(f0.z, 0.f); a[3]  = (_Float16)fmaxf(f0.w, 0.f);
        a[4]  = (_Float16)fmaxf(f1.x, 0.f); a[5]  = (_Float16)fmaxf(f1.y, 0.f);
        a[6]  = (_Float16)fmaxf(f1.z, 0.f); a[7]  = (_Float16)fmaxf(f1.w, 0.f);
        a[8]  = (_Float16)fmaxf(f2.x, 0.f); a[9]  = (_Float16)fmaxf(f2.y, 0.f);
        a[10] = (_Float16)fmaxf(f2.z, 0.f); a[11] = (_Float16)fmaxf(f2.w, 0.f);
        a[12] = (_Float16)fmaxf(f3.x, 0.f); a[13] = (_Float16)fmaxf(f3.y, 0.f);
        a[14] = (_Float16)fmaxf(f3.z, 0.f); a[15] = (_Float16)fmaxf(f3.w, 0.f);
        v16h b = *(const v16h*)(WpB + (size_t)((kt >> 5) * 4 + nt) * 512 + lane * 16);
        acc = __builtin_amdgcn_wmma_f32_16x16x32_f16(
            false, a, false, b, (short)0, acc, false, false);
    }
    #pragma unroll
    for (int v = 0; v < 8; ++v) {
        int r = m0 + v + 8 * half;
        if (r < B) g2[(size_t)r * HID + nt * 16 + ncol] = acc[v];
    }
}

// ------- GEMM2: h2lin = relu(x2) @ W2[0:64] + g2[batch]  (K=64, LDS-staged A) -------
__global__ void gemm2_wmma(const float* __restrict__ x2, const _Float16* __restrict__ Wp,
                           const float* __restrict__ g2, const int* __restrict__ batch,
                           float* __restrict__ h2lin, int N) {
    __shared__ _Float16 As[128 * LDSTR];
    int tid = threadIdx.x;
    int wave = tid >> 5, lane = tid & 31;
    int half = lane >> 4, mr = lane & 15, ncol = lane & 15;
    int blockRow0 = blockIdx.x * 128;

    v8f acc[4] = {v8f{}, v8f{}, v8f{}, v8f{}};
    #pragma unroll
    for (int kt = 0; kt < HID; kt += 32) {
        __syncthreads();
        float4 f[4]; int r_[4], c4_[4];
        #pragma unroll
        for (int i = 0; i < 4; ++i) {
            int g = tid + i * 256;
            r_[i] = g >> 3; c4_[i] = (g & 7) * 4;
            int rg = blockRow0 + r_[i]; if (rg >= N) rg = N - 1;
            float4 v = *(const float4*)(&x2[(size_t)rg * HID + kt + c4_[i]]);
            v.x = fmaxf(v.x, 0.f); v.y = fmaxf(v.y, 0.f);
            v.z = fmaxf(v.z, 0.f); v.w = fmaxf(v.w, 0.f);
            f[i] = v;
        }
        #pragma unroll
        for (int i = 0; i < 4; ++i)
            *(v4h*)(&As[r_[i] * LDSTR + c4_[i]]) = cvt4(f[i]);
        __syncthreads();
        const _Float16* ap = &As[(wave * 16 + mr) * LDSTR + 8 * half];
        v16h a = join16(*(const v8h*)ap, *(const v8h*)(ap + 16));
        int fb = (kt >> 5) * 4;
        #pragma unroll
        for (int nt = 0; nt < 4; ++nt) {
            v16h b = *(const v16h*)(Wp + (size_t)(fb + nt) * 512 + lane * 16);
            acc[nt] = __builtin_amdgcn_wmma_f32_16x16x32_f16(
                false, a, false, b, (short)0, acc[nt], false, false);
        }
    }
    int m0 = blockRow0 + wave * 16;
    #pragma unroll
    for (int v = 0; v < 8; ++v) {
        int r = m0 + v + 8 * half;
        if (r < N) {
            const float* g = g2 + (size_t)batch[r] * HID + ncol;
            float* o = h2lin + (size_t)r * HID + ncol;
            o[0]  = acc[0][v] + g[0];
            o[16] = acc[1][v] + g[16];
            o[32] = acc[2][v] + g[32];
            o[48] = acc[3][v] + g[48];
        }
    }
}

// ------- edge scatter: agg[dst] += h[src] * dinv[src]*dinv[dst]  (float4 per thread) -------
__global__ void scatter_kernel(const int* __restrict__ ei, const float* __restrict__ dinv,
                               const float* __restrict__ hin, float* __restrict__ agg, int E) {
    int t = blockIdx.x * blockDim.x + threadIdx.x;
    int e = t >> 4, j4 = (t & 15) * 4;
    if (e >= E) return;
    int s = ei[e], d = ei[E + e];
    float nrm = dinv[s] * dinv[d];
    float4 h = *(const float4*)(&hin[(size_t)s * HID + j4]);
    float* o = &agg[(size_t)d * HID + j4];
    atomicAdd(o + 0, h.x * nrm);
    atomicAdd(o + 1, h.y * nrm);
    atomicAdd(o + 2, h.z * nrm);
    atomicAdd(o + 3, h.w * nrm);
}

// ------- combine1: x2 = agg + h1/deg + b1  (float4) -------
__global__ void combine1_kernel(const float* __restrict__ agg, const float* __restrict__ h1,
                                const float* __restrict__ dinv, const float* __restrict__ b1,
                                float* __restrict__ x2, int N) {
    int t = blockIdx.x * blockDim.x + threadIdx.x;
    int n = t >> 4, j4 = (t & 15) * 4;
    if (n >= N) return;
    float di = dinv[n]; float d2 = di * di;
    float4 a = *(const float4*)(&agg[(size_t)n * HID + j4]);
    float4 h = *(const float4*)(&h1[(size_t)n * HID + j4]);
    float4 b = *(const float4*)(&b1[j4]);
    float4 r;
    r.x = a.x + h.x * d2 + b.x;
    r.y = a.y + h.y * d2 + b.y;
    r.z = a.z + h.z * d2 + b.z;
    r.w = a.w + h.w * d2 + b.w;
    *(float4*)(&x2[(size_t)n * HID + j4]) = r;
}

// ------- finalize: relu(agg2 + h2lin/deg + b2), segment-sum + counts  (float4) -------
__global__ void finalize_kernel(const float* __restrict__ agg, const float* __restrict__ h2lin,
                                const float* __restrict__ dinv, const float* __restrict__ b2,
                                const int* __restrict__ batch, float* __restrict__ outsum,
                                float* __restrict__ counts, int N) {
    int t = blockIdx.x * blockDim.x + threadIdx.x;
    int n = t >> 4, j4 = (t & 15) * 4;
    if (n >= N) return;
    float di = dinv[n]; float d2 = di * di;
    float4 a = *(const float4*)(&agg[(size_t)n * HID + j4]);
    float4 h = *(const float4*)(&h2lin[(size_t)n * HID + j4]);
    float4 b = *(const float4*)(&b2[j4]);
    int bg = batch[n];
    float* o = &outsum[(size_t)bg * HID + j4];
    atomicAdd(o + 0, fmaxf(a.x + h.x * d2 + b.x, 0.f));
    atomicAdd(o + 1, fmaxf(a.y + h.y * d2 + b.y, 0.f));
    atomicAdd(o + 2, fmaxf(a.z + h.z * d2 + b.z, 0.f));
    atomicAdd(o + 3, fmaxf(a.w + h.w * d2 + b.w, 0.f));
    if (j4 == 0) atomicAdd(&counts[bg], 1.0f);
}

// ------- writeout: out[b, 0:64] = mean; out[b, 64:128] = x2[root[b]] -------
__global__ void writeout_kernel(const float* __restrict__ outsum, const float* __restrict__ counts,
                                const float* __restrict__ x2, const int* __restrict__ rootindex,
                                float* __restrict__ out, int B) {
    int t = blockIdx.x * blockDim.x + threadIdx.x;
    if (t >= B * 128) return;
    int b = t >> 7, j = t & 127;
    if (j < HID) out[t] = outsum[(size_t)b * HID + j] / counts[b];
    else         out[t] = x2[(size_t)rootindex[b] * HID + (j - HID)];
}

extern "C" void kernel_launch(void* const* d_in, const int* in_sizes, int n_in,
                              void* d_out, int out_size, void* d_ws, size_t ws_size,
                              hipStream_t stream) {
    const float* x    = (const float*)d_in[0];
    const float* xda  = (const float*)d_in[1];
    const int*   ei   = (const int*)  d_in[2];
    const int*   batch= (const int*)  d_in[3];
    const int*   ridx = (const int*)  d_in[4];
    const float* W1   = (const float*)d_in[5];
    const float* b1   = (const float*)d_in[6];
    const float* W2   = (const float*)d_in[7];
    const float* b2   = (const float*)d_in[8];
    float* out = (float*)d_out;

    const int N = in_sizes[0] / IN1;
    const int E = in_sizes[2] / 2;
    const int B = in_sizes[4];

    float* ws = (float*)d_ws;
    size_t o = 0;
    float* deg    = ws + o; o += N;
    float* dinv   = ws + o; o += N;
    float* bufA   = ws + o; o += (size_t)N * HID;   // h1, then h2lin
    float* bufB   = ws + o; o += (size_t)N * HID;   // agg1, then agg2
    float* x2     = ws + o; o += (size_t)N * HID;
    float* g2     = ws + o; o += (size_t)B * HID;
    float* outsum = ws + o; o += (size_t)B * HID;
    float* counts = ws + o; o += B;
    o = (o + 7) & ~(size_t)7;                       // 32B-align packed weights
    _Float16* pW1  = (_Float16*)(ws + o); o += 8192;  // 8 ktiles * 4 nt * 512 halves
    _Float16* pW2a = (_Float16*)(ws + o); o += 2048;  // 2 ktiles
    _Float16* pW2b = (_Float16*)(ws + o); o += 8192;  // 8 ktiles

    const int T = 256;
    const int NH = N * HID;
    const int gblocks = (N + 127) / 128;

    // init: deg=1 (self loop), agg1=0, outsum=0, counts=0
    fillf_kernel<<<(N + T - 1) / T, T, 0, stream>>>(deg, 1.0f, N);
    fillf_kernel<<<(NH + T - 1) / T, T, 0, stream>>>(bufB, 0.0f, NH);
    fillf_kernel<<<(B * HID + T - 1) / T, T, 0, stream>>>(outsum, 0.0f, B * HID);
    fillf_kernel<<<(B + T - 1) / T, T, 0, stream>>>(counts, 0.0f, B);

    // pack weights into WMMA B-fragment order (f16)
    packW_kernel<<<(8 * 4 * 512 + T - 1) / T, T, 0, stream>>>(W1, pW1, 8, 0);
    packW_kernel<<<(2 * 4 * 512 + T - 1) / T, T, 0, stream>>>(W2, pW2a, 2, 0);
    packW_kernel<<<(8 * 4 * 512 + T - 1) / T, T, 0, stream>>>(W2, pW2b, 8, HID);

    deg_kernel<<<(E + T - 1) / T, T, 0, stream>>>(ei, deg, E);
    dinv_kernel<<<(N + T - 1) / T, T, 0, stream>>>(deg, dinv, N);

    // conv1
    gemm1_wmma<<<gblocks, T, 0, stream>>>(x, xda, pW1, bufA, N);
    scatter_kernel<<<((E * 16) + T - 1) / T, T, 0, stream>>>(ei, dinv, bufA, bufB, E);
    combine1_kernel<<<((N * 16) + T - 1) / T, T, 0, stream>>>(bufB, bufA, dinv, b1, x2, N);

    // re-zero aggregation buffer for conv2
    fillf_kernel<<<(NH + T - 1) / T, T, 0, stream>>>(bufB, 0.0f, NH);

    // conv2 linear (split: per-node 64-wide GEMM + per-graph root GEMM)
    rootgemm_wmma<<<4, T, 0, stream>>>(x, xda, pW2b, ridx, g2, B);
    gemm2_wmma<<<gblocks, T, 0, stream>>>(x2, pW2a, g2, batch, bufA, N);
    scatter_kernel<<<((E * 16) + T - 1) / T, T, 0, stream>>>(ei, dinv, bufA, bufB, E);

    // relu + segment mean + root concat
    finalize_kernel<<<((N * 16) + T - 1) / T, T, 0, stream>>>(bufB, bufA, dinv, b2, batch,
                                                              outsum, counts, N);
    writeout_kernel<<<(B * 128 + T - 1) / T, T, 0, stream>>>(outsum, counts, x2, ridx, out, B);
}
